// RQVAE_64931315581391
// MI455X (gfx1250) — compile-verified, matrix operand order checked by reading, not run on hardware
//
#include <hip/hip_runtime.h>
#include <hip/hip_bf16.h>

// ---------------------------------------------------------------------------
// RQ-VAE forward for MI455X (gfx1250, wave32).
// GEMMs: bf16 v_wmma_f32_16x16x32_bf16, tiles staged by the Tensor Data Mover
// (tensor_load_to_lds + s_wait_tensorcnt), double-buffered LDS.
// LayerNorm + 3-stage residual VQ fused in one wave-per-row f32 kernel.
// ---------------------------------------------------------------------------

typedef __bf16 bf16_t;
typedef __attribute__((ext_vector_type(16))) __bf16 v16bf;
typedef __attribute__((ext_vector_type(8)))  __bf16 bf16x8;
typedef __attribute__((ext_vector_type(4)))  __bf16 bf16x4;
typedef __attribute__((ext_vector_type(8)))  float   v8f;
typedef __attribute__((ext_vector_type(4)))  unsigned int u32x4;
typedef __attribute__((ext_vector_type(8)))  unsigned int u32x8;

static constexpr int   B_    = 16384;
static constexpr int   DIN   = 2048;
static constexpr int   H1_   = 1024;
static constexpr int   H2_   = 512;
static constexpr int   DL    = 256;
static constexpr int   NCODE = 1024;
static constexpr float BETA_ = 0.25f;

// LDS geometry for the TDM-fed GEMM: rows are 32 bf16 (64 B) + 16 B pad = 80 B
// pitch (20-bank stride -> conflict-free b128 fragment loads).
static constexpr unsigned ROW_B     = 80;                 // bytes per LDS row
static constexpr unsigned A_ROWS    = 128;
static constexpr unsigned B_ROWS    = 256;
static constexpr unsigned B_OFF     = A_ROWS * ROW_B;     // 10240
static constexpr unsigned BUF_BYTES = (A_ROWS + B_ROWS) * ROW_B;  // 30720

// ---------------------------- f32 -> bf16 ----------------------------------
__global__ __launch_bounds__(256) void cvt_bf16_kernel(const float* __restrict__ in,
                                                       bf16_t* __restrict__ out, int n4) {
  int i = blockIdx.x * blockDim.x + threadIdx.x;
  if (i < n4) {
    float4 v = ((const float4*)in)[i];
    bf16x4 o;
    o[0] = (bf16_t)v.x; o[1] = (bf16_t)v.y; o[2] = (bf16_t)v.z; o[3] = (bf16_t)v.w;
    ((bf16x4*)out)[i] = o;
  }
}

// --------------- W[K,N] f32  ->  Wt[N,K] bf16 (LDS tiled) ------------------
__global__ __launch_bounds__(256) void transpose_bf16_kernel(const float* __restrict__ W,
                                                             bf16_t* __restrict__ Wt,
                                                             int K, int N) {
  __shared__ float tile[32][33];
  const int n0 = blockIdx.x * 32, k0 = blockIdx.y * 32;
  const int tx = threadIdx.x, ty = threadIdx.y;       // blockDim = (32, 8)
#pragma unroll
  for (int i = 0; i < 32; i += 8)
    tile[ty + i][tx] = W[(size_t)(k0 + ty + i) * N + (n0 + tx)];
  __syncthreads();
#pragma unroll
  for (int i = 0; i < 32; i += 8)
    Wt[(size_t)(n0 + ty + i) * K + (k0 + tx)] = (bf16_t)tile[tx][ty + i];
}

// ------------------------- ||cb_k||^2 precompute ---------------------------
__global__ __launch_bounds__(256) void cb_norm_kernel(const float* __restrict__ cb,
                                                      float* __restrict__ out) {
  const int lane = threadIdx.x & 31, wave = threadIdx.x >> 5;
  const int c = blockIdx.x * 8 + wave;
  const float4* cv = (const float4*)(cb + (size_t)c * DL);
  float4 a = cv[lane * 2], b = cv[lane * 2 + 1];
  float s = a.x*a.x + a.y*a.y + a.z*a.z + a.w*a.w
          + b.x*b.x + b.y*b.y + b.z*b.z + b.w*b.w;
#pragma unroll
  for (int m = 16; m >= 1; m >>= 1) s += __shfl_xor(s, m, 32);
  if (lane == 0) out[c] = s;
}

// ------------------- Tensor Data Mover: 2-D bf16 tile load -----------------
// D# group0/group1 per CDNA5 ISA 8.3/8.4. 2-D tensor (groups 2/3 disabled via
// the 2-operand form). Tile = 32 elements (64 B) x tileRows, row pitch K elems
// in memory; LDS gets 16 B pad after every 64 B -> 80 B LDS pitch.
__device__ __forceinline__ void tdm_load_tile(unsigned ldsOff, unsigned long long gaddr,
                                              unsigned K_elems, unsigned rowsTot,
                                              unsigned tileRows) {
  u32x4 g0;
  g0[0] = 1u;                                                   // count=1 (valid)
  g0[1] = ldsOff;                                               // lds_addr (bytes)
  g0[2] = (unsigned)gaddr;                                      // global_addr lo
  g0[3] = ((unsigned)(gaddr >> 32) & 0x01FFFFFFu) | 0x80000000u; // addr[56:32] | type=2
  u32x8 g1;
  g1[0] = (1u << 16)        // data_size = 2 bytes
        | (1u << 20)        // pad_enable
        | (3u << 22)        // pad_interval: every 16 DWORDs (64 B)
        | (3u << 25);       // pad_amount: 4 DWORDs (16 B)
  g1[1] = (K_elems & 0xFFFFu) << 16;                            // tensor_dim0 lo16
  g1[2] = ((K_elems >> 16) & 0xFFFFu) | ((rowsTot & 0xFFFFu) << 16);
  g1[3] = ((rowsTot >> 16) & 0xFFFFu) | (32u << 16);            // tensor_dim1 hi | tile_dim0=32
  g1[4] = tileRows & 0xFFFFu;                                   // tile_dim1 | tile_dim2=0
  g1[5] = K_elems;                                              // tensor_dim0_stride lo32
  g1[6] = 0u;                                                   // stride hi | dim1_stride lo
  g1[7] = 0u;
  asm volatile("tensor_load_to_lds %0, %1" :: "s"(g0), "s"(g1) : "memory");
}

// --------------------------- WMMA fragment load ----------------------------
// 16-bit A/B fragment, wave32: lane = row (M or N), elements 0..7 hold K
// {0..7 | lane<16 ; 8..15 | lane>=16}, elements 8..15 hold K {+16}.
__device__ __forceinline__ v16bf load_frag(const bf16_t* rowp, int hi) {
  bf16x8 a = *(const bf16x8*)(rowp + 8 * hi);        // 16B ds_load
  bf16x8 b = *(const bf16x8*)(rowp + 16 + 8 * hi);   // 16B ds_load
  v16bf r;
#pragma unroll
  for (int i = 0; i < 8; ++i) { r[i] = a[i]; r[i + 8] = b[i]; }
  return r;
}

// ------ GEMM: Out[M,N] = act(A[M,K] @ Wt[N,K]^T + bias), TDM + bf16 WMMA ---
// Block tile 128x256, BK=32, 8 waves (2x4), each wave a 64x64 C tile
// (4x4 WMMA tiles = 16 v_wmma per K-step). Waves 0/1 drive the TDM (A / B
// tiles), double-buffered; compute waves only touch LDS + WMMA.
template <typename OutT, bool RELU>
__global__ __launch_bounds__(256) void gemm_tdm_kernel(const bf16_t* __restrict__ A,
                                                       const bf16_t* __restrict__ Wt,
                                                       const float* __restrict__ bias,
                                                       OutT* __restrict__ Out,
                                                       int M, int N, int K) {
  __shared__ __align__(128) unsigned char smem[2 * BUF_BYTES];

  const int lane = threadIdx.x & 31;
  const int wave = threadIdx.x >> 5;
  const int wm = wave >> 2;          // 0..1 (M)
  const int wn = wave & 3;           // 0..3 (N)
  const int m0 = blockIdx.y * 128;
  const int n0 = blockIdx.x * 256;
  const int hi = lane >> 4;
  const int lr = lane & 15;

  const unsigned ldsBase = (unsigned)(unsigned long long)(unsigned char*)smem;

  // Waves 0 and 1 are the tile loaders (A and B respectively).
  const bool loader = (wave < 2);
  unsigned long long gBase = 0; unsigned rowsTot = 0, tileRows = 0, regionOff = 0;
  if (wave == 0) {
    gBase = (unsigned long long)(const void*)A + (unsigned long long)m0 * K * 2;
    rowsTot = (unsigned)M; tileRows = A_ROWS; regionOff = 0;
  } else if (wave == 1) {
    gBase = (unsigned long long)(const void*)Wt + (unsigned long long)n0 * K * 2;
    rowsTot = (unsigned)N; tileRows = B_ROWS; regionOff = B_OFF;
  }

  v8f acc[4][4];
#pragma unroll
  for (int mt = 0; mt < 4; ++mt)
#pragma unroll
    for (int nt = 0; nt < 4; ++nt)
#pragma unroll
      for (int r = 0; r < 8; ++r) acc[mt][nt][r] = 0.f;

  const int nk = K / 32;
  if (loader)
    tdm_load_tile(ldsBase + regionOff, gBase, (unsigned)K, rowsTot, tileRows);

  for (int kt = 0; kt < nk; ++kt) {
    const int cur = kt & 1;
    if (loader) {
      if (kt + 1 < nk) {
        tdm_load_tile(ldsBase + (cur ^ 1) * BUF_BYTES + regionOff,
                      gBase + (unsigned long long)(kt + 1) * 64u,
                      (unsigned)K, rowsTot, tileRows);
        __builtin_amdgcn_s_wait_tensorcnt(1);   // current tile has landed
      } else {
        __builtin_amdgcn_s_wait_tensorcnt(0);
      }
    }
    __syncthreads();                            // tile visible to all waves

    const bf16_t* Ab = (const bf16_t*)(smem + cur * BUF_BYTES);
    const bf16_t* Bb = (const bf16_t*)(smem + cur * BUF_BYTES + B_OFF);

    v16bf bfr[4];
#pragma unroll
    for (int nt = 0; nt < 4; ++nt)
      bfr[nt] = load_frag(Bb + (unsigned)(wn * 64 + nt * 16 + lr) * (ROW_B / 2), hi);
#pragma unroll
    for (int mt = 0; mt < 4; ++mt) {
      v16bf af = load_frag(Ab + (unsigned)(wm * 64 + mt * 16 + lr) * (ROW_B / 2), hi);
#pragma unroll
      for (int nt = 0; nt < 4; ++nt)
        acc[mt][nt] = __builtin_amdgcn_wmma_f32_16x16x32_bf16(
            false, af, false, bfr[nt], (short)0, acc[mt][nt], false, false);
    }
    __syncthreads();                            // done reading before refill
  }

  // Epilogue: C layout — VGPR r holds row (r + 8*hi), lane lr holds col.
#pragma unroll
  for (int nt = 0; nt < 4; ++nt) {
    const int col = n0 + wn * 64 + nt * 16 + lr;
    const float bv = bias[col];
#pragma unroll
    for (int mt = 0; mt < 4; ++mt) {
#pragma unroll
      for (int r = 0; r < 8; ++r) {
        float v = acc[mt][nt][r] + bv;
        if (RELU) v = fmaxf(v, 0.f);
        const int row = m0 + wm * 64 + mt * 16 + r + 8 * hi;
        Out[(size_t)row * N + col] = (OutT)v;
      }
    }
  }
}

// ------------- fused LayerNorm + 3-stage residual VQ (wave/row) ------------
__global__ __launch_bounds__(256) void vq_fused_kernel(
    const float* __restrict__ Z,
    const float* __restrict__ ln_g, const float* __restrict__ ln_b,
    const float* __restrict__ cb0, const float* __restrict__ cb1,
    const float* __restrict__ cb2, const float* __restrict__ cbn,
    bf16_t* __restrict__ Zq, int* __restrict__ IdxOut, float* __restrict__ VqLoss) {
  const int lane = threadIdx.x & 31;
  const int wave = threadIdx.x >> 5;
  const int b = blockIdx.x * 8 + wave;

  const float4* zv = (const float4*)(Z + (size_t)b * DL);
  float4 a0 = zv[lane * 2], a1 = zv[lane * 2 + 1];
  float zr[8] = {a0.x, a0.y, a0.z, a0.w, a1.x, a1.y, a1.z, a1.w};

  float s1 = 0.f, s2 = 0.f;
#pragma unroll
  for (int j = 0; j < 8; ++j) { s1 += zr[j]; s2 += zr[j] * zr[j]; }
#pragma unroll
  for (int m = 16; m >= 1; m >>= 1) { s1 += __shfl_xor(s1, m, 32); s2 += __shfl_xor(s2, m, 32); }
  const float mu  = s1 * (1.f / DL);
  const float var = s2 * (1.f / DL) - mu * mu;
  const float inv = rsqrtf(var + 1e-5f);

  float4 g0 = ((const float4*)ln_g)[lane * 2], g1 = ((const float4*)ln_g)[lane * 2 + 1];
  float4 q0 = ((const float4*)ln_b)[lane * 2], q1 = ((const float4*)ln_b)[lane * 2 + 1];
  float gg[8] = {g0.x, g0.y, g0.z, g0.w, g1.x, g1.y, g1.z, g1.w};
  float bb[8] = {q0.x, q0.y, q0.z, q0.w, q1.x, q1.y, q1.z, q1.w};

  float r[8], zq[8];
#pragma unroll
  for (int j = 0; j < 8; ++j) { r[j] = (zr[j] - mu) * inv * gg[j] + bb[j]; zq[j] = 0.f; }

  float lossAcc = 0.f;
  const float* cbs[3] = {cb0, cb1, cb2};
#pragma unroll 1
  for (int st = 0; st < 3; ++st) {
    const float* cb = cbs[st];
    const float* cn = cbn + st * NCODE;

    float rn = 0.f;
#pragma unroll
    for (int j = 0; j < 8; ++j) rn += r[j] * r[j];
#pragma unroll
    for (int m = 16; m >= 1; m >>= 1) rn += __shfl_xor(rn, m, 32);

    float best = 3.402823e38f; int bidx = 0;
    for (int k = 0; k < NCODE; ++k) {
      const float4* cv = (const float4*)(cb + (size_t)k * DL);
      float4 c0 = cv[lane * 2], c1 = cv[lane * 2 + 1];
      float p = r[0]*c0.x + r[1]*c0.y + r[2]*c0.z + r[3]*c0.w
              + r[4]*c1.x + r[5]*c1.y + r[6]*c1.z + r[7]*c1.w;
#pragma unroll
      for (int m = 16; m >= 1; m >>= 1) p += __shfl_xor(p, m, 32);
      float d = cn[k] - 2.f * p;          // rn omitted: constant per row
      if (d < best) { best = d; bidx = k; }
    }
    lossAcc += (1.f + BETA_) * (rn + best) * (1.f / DL);

    const float4* ev = (const float4*)(cb + (size_t)bidx * DL);
    float4 e0 = ev[lane * 2], e1 = ev[lane * 2 + 1];
    float ee[8] = {e0.x, e0.y, e0.z, e0.w, e1.x, e1.y, e1.z, e1.w};
#pragma unroll
    for (int j = 0; j < 8; ++j) { r[j] -= ee[j]; zq[j] += ee[j]; }
    if (lane == 0) IdxOut[b * 3 + st] = bidx;
  }

  bf16x8 o;
#pragma unroll
  for (int j = 0; j < 8; ++j) o[j] = (bf16_t)zq[j];
  ((bf16x8*)(Zq + (size_t)b * DL))[lane] = o;
  if (lane == 0) VqLoss[b] = lossAcc;
}

// --------------------- recon loss + total loss (block/row) -----------------
__global__ __launch_bounds__(256) void recon_loss_kernel(const float* __restrict__ Xhat,
                                                         const float* __restrict__ X,
                                                         const float* __restrict__ VqLoss,
                                                         float* __restrict__ LossOut) {
  __shared__ float wsum[8];
  const int b = blockIdx.x;
  const float* xh = Xhat + (size_t)b * DIN;
  const float* xr = X    + (size_t)b * DIN;
  float s = 0.f;
  for (int i = threadIdx.x; i < DIN; i += 256) {
    float d = xh[i] - xr[i];
    s += d * d;
  }
#pragma unroll
  for (int m = 16; m >= 1; m >>= 1) s += __shfl_xor(s, m, 32);
  if ((threadIdx.x & 31) == 0) wsum[threadIdx.x >> 5] = s;
  __syncthreads();
  if (threadIdx.x == 0) {
    float t = 0.f;
#pragma unroll
    for (int i = 0; i < 8; ++i) t += wsum[i];
    LossOut[b] = VqLoss[b] + t * (1.f / DIN);
  }
}

// ---------------------------------------------------------------------------
extern "C" void kernel_launch(void* const* d_in, const int* in_sizes, int n_in,
                              void* d_out, int out_size, void* d_ws, size_t ws_size,
                              hipStream_t stream) {
  (void)in_sizes; (void)n_in; (void)out_size; (void)ws_size;
  const float* x   = (const float*)d_in[0];
  const float* eW1 = (const float*)d_in[1];
  const float* eb1 = (const float*)d_in[2];
  const float* eW2 = (const float*)d_in[3];
  const float* eb2 = (const float*)d_in[4];
  const float* eW3 = (const float*)d_in[5];
  const float* eb3 = (const float*)d_in[6];
  const float* lng = (const float*)d_in[7];
  const float* lnb = (const float*)d_in[8];
  const float* cb0 = (const float*)d_in[9];
  const float* cb1 = (const float*)d_in[10];
  const float* cb2 = (const float*)d_in[11];
  const float* dW1 = (const float*)d_in[12];
  const float* db1 = (const float*)d_in[13];
  const float* dW2 = (const float*)d_in[14];
  const float* db2 = (const float*)d_in[15];
  const float* dW3 = (const float*)d_in[16];
  const float* db3 = (const float*)d_in[17];

  // Output layout: x_hat[f32 B*DIN] | loss[f32 B] | idx[i32 B*3]
  float* xhat    = (float*)d_out;
  float* lossOut = xhat + (size_t)B_ * DIN;
  int*   idxOut  = (int*)(lossOut + B_);

  // Workspace carve-up (all chunk sizes multiples of 256 B).
  char* ws = (char*)d_ws;
  size_t off = 0;
  auto carve = [&](size_t bytes) { char* p = ws + off; off += bytes; return p; };
  bf16_t* xb    = (bf16_t*)carve((size_t)B_ * DIN * 2);
  bf16_t* h1b   = (bf16_t*)carve((size_t)B_ * H1_ * 2);
  bf16_t* h2b   = (bf16_t*)carve((size_t)B_ * H2_ * 2);
  float*  zf    = (float*) carve((size_t)B_ * DL  * 4);
  bf16_t* zqb   = (bf16_t*)carve((size_t)B_ * DL  * 2);
  bf16_t* d1b   = (bf16_t*)carve((size_t)B_ * H2_ * 2);
  bf16_t* d2b   = (bf16_t*)carve((size_t)B_ * H1_ * 2);
  bf16_t* wtE1  = (bf16_t*)carve((size_t)H1_ * DIN * 2);
  bf16_t* wtE2  = (bf16_t*)carve((size_t)H2_ * H1_ * 2);
  bf16_t* wtE3  = (bf16_t*)carve((size_t)DL  * H2_ * 2);
  bf16_t* wtD1  = (bf16_t*)carve((size_t)H2_ * DL  * 2);
  bf16_t* wtD2  = (bf16_t*)carve((size_t)H1_ * H2_ * 2);
  bf16_t* wtD3  = (bf16_t*)carve((size_t)DIN * H1_ * 2);
  float*  cbn   = (float*) carve((size_t)3 * NCODE * 4);
  float*  vqls  = (float*) carve((size_t)B_ * 4);

  const dim3 tb(32, 8);

  // 1) precision conversion / weight transposition
  cvt_bf16_kernel<<<(B_ * DIN / 4 + 255) / 256, 256, 0, stream>>>(x, xb, B_ * DIN / 4);
  transpose_bf16_kernel<<<dim3(H1_ / 32, DIN / 32), tb, 0, stream>>>(eW1, wtE1, DIN, H1_);
  transpose_bf16_kernel<<<dim3(H2_ / 32, H1_ / 32), tb, 0, stream>>>(eW2, wtE2, H1_, H2_);
  transpose_bf16_kernel<<<dim3(DL  / 32, H2_ / 32), tb, 0, stream>>>(eW3, wtE3, H2_, DL);
  transpose_bf16_kernel<<<dim3(H2_ / 32, DL  / 32), tb, 0, stream>>>(dW1, wtD1, DL,  H2_);
  transpose_bf16_kernel<<<dim3(H1_ / 32, H2_ / 32), tb, 0, stream>>>(dW2, wtD2, H2_, H1_);
  transpose_bf16_kernel<<<dim3(DIN / 32, H1_ / 32), tb, 0, stream>>>(dW3, wtD3, H1_, DIN);
  cb_norm_kernel<<<NCODE / 8, 256, 0, stream>>>(cb0, cbn);
  cb_norm_kernel<<<NCODE / 8, 256, 0, stream>>>(cb1, cbn + NCODE);
  cb_norm_kernel<<<NCODE / 8, 256, 0, stream>>>(cb2, cbn + 2 * NCODE);

  // 2) encoder (TDM-fed bf16 WMMA, fused bias+ReLU; last layer f32, no act)
  gemm_tdm_kernel<bf16_t, true ><<<dim3(H1_ / 256, B_ / 128), 256, 0, stream>>>(xb,  wtE1, eb1, h1b, B_, H1_, DIN);
  gemm_tdm_kernel<bf16_t, true ><<<dim3(H2_ / 256, B_ / 128), 256, 0, stream>>>(h1b, wtE2, eb2, h2b, B_, H2_, H1_);
  gemm_tdm_kernel<float,  false><<<dim3(DL  / 256, B_ / 128), 256, 0, stream>>>(h2b, wtE3, eb3, zf,  B_, DL,  H2_);

  // 3) LayerNorm + residual VQ (3 codebooks), fused
  vq_fused_kernel<<<B_ / 8, 256, 0, stream>>>(zf, lng, lnb, cb0, cb1, cb2, cbn,
                                              zqb, idxOut, vqls);

  // 4) decoder
  gemm_tdm_kernel<bf16_t, true ><<<dim3(H2_ / 256, B_ / 128), 256, 0, stream>>>(zqb, wtD1, db1, d1b, B_, H2_, DL);
  gemm_tdm_kernel<bf16_t, true ><<<dim3(H1_ / 256, B_ / 128), 256, 0, stream>>>(d1b, wtD2, db2, d2b, B_, H1_, H2_);
  gemm_tdm_kernel<float,  false><<<dim3(DIN / 256, B_ / 128), 256, 0, stream>>>(d2b, wtD3, db3, xhat, B_, DIN, H1_);

  // 5) recon + total loss
  recon_loss_kernel<<<B_, 256, 0, stream>>>(xhat, x, vqls, lossOut);
}
